// Mamba_17557826306685
// MI455X (gfx1250) — matrix operand order repaired and checked by reading, not compiled, not run on hardware
//
#include <hip/hip_runtime.h>
#include <hip/hip_bf16.h>

// ---------------------------------------------------------------------------
// Mamba block for MI455X (gfx1250): WMMA bf16 GEMMs with async global->LDS
// weight staging + wave32 parallel selective scan.
// ---------------------------------------------------------------------------

#define BATCH   4
#define LSEQ    2560
#define DM      128
#define DI      256
#define DSTATE  16
#define DTRANK  8
#define TT      160

typedef __bf16 v16bf __attribute__((ext_vector_type(16)));
typedef float  v8f   __attribute__((ext_vector_type(8)));

union Frag32B { uint4 q[2]; v16bf v; };

// ----------------------------- WMMA fragment helpers -----------------------
// A tile in LDS: row-major [64][32] bf16. Lane layout (ISA 7.12.2):
//   lanes 0-15: M=lane,   elems {0..7}=K(8hi+0..7) with hi=0 -> K0..7,
//   elems {8..15}=K16..23 ; lanes 16-31 shift K by 8.
// => two contiguous 16B runs per lane: ds_load_b128 x2.
__device__ __forceinline__ v16bf load_a_frag(const __bf16* As, int mBase, int lane) {
  int m  = mBase + (lane & 15);
  int hi = lane >> 4;
  const uint4* p = (const uint4*)(As + m * 32);
  Frag32B f;
  f.q[0] = p[hi];       // K = 8hi .. 8hi+7
  f.q[1] = p[2 + hi];   // K = 16+8hi .. 16+8hi+7
  return f.v;
}
// B tile in LDS: n-major [BN][32] bf16 (Bs[n*32+k]). Lane layout:
//   lane = N + 16*hi, elems {0..7}=K(8hi+0..7), elems {8..15}=K(16+8hi+0..7).
__device__ __forceinline__ v16bf load_b_frag(const __bf16* Bs, int nBase, int lane) {
  int n  = nBase + (lane & 15);
  int hi = lane >> 4;
  const uint4* p = (const uint4*)(Bs + n * 32);
  Frag32B f;
  f.q[0] = p[hi];
  f.q[1] = p[2 + hi];
  return f.v;
}
// C/D (16x16 f32): vgpr r: lanes0-15 M=r, lanes16-31 M=r+8; N = lane&15
__device__ __forceinline__ void store_c(float* C, int ldc, int row0, int col0,
                                        int lane, const v8f& c) {
  int col   = col0 + (lane & 15);
  int rbase = row0 + 8 * (lane >> 4);
#pragma unroll
  for (int r = 0; r < 8; ++r) C[(rbase + r) * ldc + col] = c[r];
}

__device__ __forceinline__ unsigned pack_bf16(float a, float b) {
  unsigned short ua = __builtin_bit_cast(unsigned short, (__bf16)a);
  unsigned short ub = __builtin_bit_cast(unsigned short, (__bf16)b);
  return (unsigned)ua | ((unsigned)ub << 16);
}

// wave-relative LDS byte offset = low 32 bits of generic shared pointer
__device__ __forceinline__ unsigned lds_offset(const void* p) {
  return (unsigned)(unsigned long long)p;
}

// async copy of `chunks` 16-byte blocks global->LDS, cooperatively
__device__ __forceinline__ void async_copy_tile(const void* gsrc, const void* ldst,
                                                int chunks, int tid, int nthr) {
  unsigned long long g = (unsigned long long)gsrc;
  unsigned l = lds_offset(ldst);
  for (int i = tid; i < chunks; i += nthr) {
    unsigned long long ga = g + (unsigned long long)i * 16ull;
    unsigned la = l + (unsigned)i * 16u;
    asm volatile("global_load_async_to_lds_b128 %0, %1, off"
                 :: "v"(la), "v"(ga) : "memory");
  }
  asm volatile("s_wait_asynccnt 0x0" ::: "memory");
}

// ----------------------------- stage 0: front end --------------------------
__global__ void pool_kernel(const float* __restrict__ x, float* __restrict__ xo) {
  int idx = blockIdx.x * blockDim.x + threadIdx.x;   // B*128*16 = 8192
  if (idx >= BATCH * DM * 16) return;
  int p = idx & 15, dm = (idx >> 4) & 127, b = idx >> 11;
  const float* xp = x + ((long)b * LSEQ + p) * DM + dm;
  float s = 0.f, mx = -1e30f;
  for (int t = 0; t < TT; ++t) {
    float v = xp[(long)t * 16 * DM];
    s += v; mx = fmaxf(mx, v);
  }
  xo[(b * 2 * DM + dm) * 16 + p]      = s * (1.f / (float)TT);
  xo[(b * 2 * DM + DM + dm) * 16 + p] = mx;
}

__global__ void dwconv_kernel(const float* __restrict__ xo,
                              const float* __restrict__ w,
                              float* __restrict__ xc1) {
  int idx = blockIdx.x * blockDim.x + threadIdx.x;   // 8192
  if (idx >= BATCH * DM * 16) return;
  int p = idx & 15, oc = (idx >> 4) & 127, b = idx >> 11;
  int j = p & 3, i = p >> 2;
  float s = 0.f;
#pragma unroll
  for (int ic = 0; ic < 2; ++ic) {
    const float* src = xo + (b * 2 * DM + 2 * oc + ic) * 16;
    const float* wp  = w + (oc * 2 + ic) * 9;
#pragma unroll
    for (int dy = 0; dy < 3; ++dy)
#pragma unroll
      for (int dx = 0; dx < 3; ++dx) {
        int yy = i + dy - 1, xx = j + dx - 1;
        if (yy >= 0 && yy < 4 && xx >= 0 && xx < 4)
          s += wp[dy * 3 + dx] * src[yy * 4 + xx];
      }
  }
  xc1[idx] = (s >= 0.f) ? s : 0.1f * s;
}

__global__ void offconv_kernel(const float* __restrict__ xc1,
                               const float* __restrict__ w,
                               float* __restrict__ offmap) {
  int idx = threadIdx.x;                              // 128 = B*2*16
  if (idx >= BATCH * 2 * 16) return;
  int p = idx & 15, o = (idx >> 4) & 1, b = idx >> 5;
  int j = p & 3, i = p >> 2;
  float s = 0.f;
  for (int c = 0; c < DM; ++c) {
    const float* src = xc1 + (b * DM + c) * 16;
    const float* wp  = w + (o * DM + c) * 9;
#pragma unroll
    for (int dy = 0; dy < 3; ++dy)
#pragma unroll
      for (int dx = 0; dx < 3; ++dx) {
        int yy = i + dy - 1, xx = j + dx - 1;
        if (yy >= 0 && yy < 4 && xx >= 0 && xx < 4)
          s += wp[dy * 3 + dx] * src[yy * 4 + xx];
      }
  }
  offmap[b * 32 + o * 16 + p] = s;
}

__global__ void warp_kernel(const float* __restrict__ x,
                            const float* __restrict__ offmap,
                            float* __restrict__ xr) {
  long idx = (long)blockIdx.x * blockDim.x + threadIdx.x;  // 4*160*16*128
  if (idx >= (long)BATCH * TT * 16 * DM) return;
  int dm = idx & 127;
  long r = idx >> 7;
  int p = r & 15; r >>= 4;
  int t = (int)(r % TT);
  int b = (int)(r / TT);
  int j = p & 3, i = p >> 2;
  float ox = offmap[b * 32 + p];
  float oy = offmap[b * 32 + 16 + p];
  float px = fminf(fmaxf((float)j + ox, 0.f), 3.f);
  float py = fminf(fmaxf((float)i + oy, 0.f), 3.f);
  float x0f = floorf(px), y0f = floorf(py);
  float wx = px - x0f, wy = py - y0f;
  int x0 = (int)x0f, y0 = (int)y0f;
  int x1 = (x0 + 1 < 3) ? x0 + 1 : 3;
  int y1 = (y0 + 1 < 3) ? y0 + 1 : 3;
  const float* xb = x + ((long)b * LSEQ + t * 16) * DM + dm;
  float v00 = xb[(y0 * 4 + x0) * DM];
  float v01 = xb[(y0 * 4 + x1) * DM];
  float v10 = xb[(y1 * 4 + x0) * DM];
  float v11 = xb[(y1 * 4 + x1) * DM];
  float warped = v00 * (1.f - wx) * (1.f - wy) + v01 * wx * (1.f - wy)
               + v10 * (1.f - wx) * wy         + v11 * wx * wy;
  xr[((long)b * LSEQ + t * 16 + p) * DM + dm] = xb[p * DM] + warped;
}

// ------------------- weight conversion into tiled n-major LDS images -------
// W_in (512x128): 4 nBlocks x 4 kSteps, tile [128][32]: idx ((nb*4+ks)*128+n)*32+k
// W_xproj padded (48x256): 8 kSteps, tile [48][32]
// W_out (128x256): 8 kSteps, tile [128][32]
__global__ void convert_weights(const float* __restrict__ Win,
                                const float* __restrict__ Wxp,
                                const float* __restrict__ Wout,
                                unsigned short* WinB_, unsigned short* WxpB_,
                                unsigned short* WoutB_) {
  __bf16* WinB  = (__bf16*)WinB_;
  __bf16* WxpB  = (__bf16*)WxpB_;
  __bf16* WoutB = (__bf16*)WoutB_;
  int i = blockIdx.x * blockDim.x + threadIdx.x;
  if (i < 2 * DI * DM) {                         // 65536
    int k = i & 31, n = (i >> 5) & 127, ks = (i >> 12) & 3, nb = i >> 14;
    WinB[i] = (__bf16)Win[(nb * 128 + n) * DM + ks * 32 + k];
  }
  if (i < 48 * DI) {                             // 12288
    int k = i & 31, n = (i >> 5) % 48, ks = i / 1536;
    WxpB[i] = (n < DTRANK + 2 * DSTATE) ? (__bf16)Wxp[n * DI + ks * 32 + k]
                                        : (__bf16)0.f;
  }
  if (i < DM * DI) {                             // 32768
    int k = i & 31, n = (i >> 5) & 127, ks = i >> 12;
    WoutB[i] = (__bf16)Wout[n * DI + ks * 32 + k];
  }
}

// ----------------------------- GEMM 1: xz = xr @ W_in^T --------------------
// M=10240 N=512 K=128, BM=64 BN=128, 8 waves (4 M-sub x 2 N-groups of 4)
__global__ __launch_bounds__(256) void gemm_xin(const float* __restrict__ Xr,
                                                const unsigned short* __restrict__ Wb_,
                                                float* __restrict__ XZ) {
  const __bf16* Wb = (const __bf16*)Wb_;
  __shared__ alignas(16) __bf16 As[64 * 32];
  __shared__ alignas(16) __bf16 Bs[128 * 32];
  int tid = threadIdx.x, wave = tid >> 5, lane = tid & 31;
  int mBlock = blockIdx.x, nBlock = blockIdx.y;
  int mSub = (wave & 3) * 16;
  int nGrp = (wave >> 2) * 64;
  v8f acc[4] = {};
  for (int ks = 0; ks < 4; ++ks) {
    // A fill: packed bf16 pairs (64 rows x 16 dwords)
    unsigned* Ap = (unsigned*)As;
    for (int i = tid; i < 64 * 16; i += 256) {
      int r = i >> 4, kk = (i & 15) * 2;
      const float2 v = *(const float2*)&Xr[(mBlock * 64 + r) * DM + ks * 32 + kk];
      Ap[i] = pack_bf16(v.x, v.y);
    }
    // B tile: straight async global->LDS copy (pre-tiled n-major image)
    async_copy_tile(Wb + (nBlock * 4 + ks) * (128 * 32), Bs, 128 * 4, tid, 256);
    __syncthreads();
    v16bf a = load_a_frag(As, mSub, lane);
#pragma unroll
    for (int j = 0; j < 4; ++j) {
      v16bf b = load_b_frag(Bs, nGrp + j * 16, lane);
      acc[j] = __builtin_amdgcn_wmma_f32_16x16x32_bf16(
          false, a, false, b, (short)0, acc[j], false, false);
    }
    __syncthreads();
  }
#pragma unroll
  for (int j = 0; j < 4; ++j)
    store_c(XZ, 2 * DI, mBlock * 64 + mSub, nBlock * 128 + nGrp + j * 16, lane, acc[j]);
}

// ----------------------------- depthwise conv1d + SiLU ---------------------
__global__ void conv1d_kernel(const float* __restrict__ xz,
                              const float* __restrict__ w,
                              const float* __restrict__ bias,
                              float* __restrict__ xm) {
  long idx = (long)blockIdx.x * blockDim.x + threadIdx.x;   // 10240*256
  if (idx >= (long)BATCH * LSEQ * DI) return;
  int c = idx & 255;
  long bl = idx >> 8;
  int l = (int)(bl % LSEQ);
  float s = bias[c];
#pragma unroll
  for (int k = 0; k < 3; ++k) {
    int ll = l + k - 1;
    if (ll >= 0 && ll < LSEQ)
      s += w[c * 3 + k] * xz[(bl + (k - 1)) * (2 * DI) + c];
  }
  xm[bl * DI + c] = s / (1.f + __expf(-s));
}

// ----------------------------- GEMM 2: dBC = xm @ W_xproj^T (N pad 48) -----
__global__ __launch_bounds__(128) void gemm_xproj(const float* __restrict__ Xm,
                                                  const unsigned short* __restrict__ Wb_,
                                                  float* __restrict__ dBC) {
  const __bf16* Wb = (const __bf16*)Wb_;
  __shared__ alignas(16) __bf16 As[64 * 32];
  __shared__ alignas(16) __bf16 Bs[48 * 32];
  int tid = threadIdx.x, wave = tid >> 5, lane = tid & 31;
  int mBlock = blockIdx.x;
  int mSub = wave * 16;
  v8f acc[3] = {};
  for (int ks = 0; ks < 8; ++ks) {
    unsigned* Ap = (unsigned*)As;
    for (int i = tid; i < 64 * 16; i += 128) {
      int r = i >> 4, kk = (i & 15) * 2;
      const float2 v = *(const float2*)&Xm[(mBlock * 64 + r) * DI + ks * 32 + kk];
      Ap[i] = pack_bf16(v.x, v.y);
    }
    async_copy_tile(Wb + ks * (48 * 32), Bs, 48 * 4, tid, 128);
    __syncthreads();
    v16bf a = load_a_frag(As, mSub, lane);
#pragma unroll
    for (int j = 0; j < 3; ++j) {
      v16bf b = load_b_frag(Bs, j * 16, lane);
      acc[j] = __builtin_amdgcn_wmma_f32_16x16x32_bf16(
          false, a, false, b, (short)0, acc[j], false, false);
    }
    __syncthreads();
  }
#pragma unroll
  for (int j = 0; j < 3; ++j)
    store_c(dBC, 48, mBlock * 64 + mSub, j * 16, lane, acc[j]);
}

// ----------------------------- dt projection + softplus --------------------
__global__ void dt_kernel(const float* __restrict__ dBC,
                          const float* __restrict__ Wdt,
                          const float* __restrict__ bdt,
                          float* __restrict__ dlt) {
  long idx = (long)blockIdx.x * blockDim.x + threadIdx.x;   // 10240*256
  if (idx >= (long)BATCH * LSEQ * DI) return;
  int c = idx & 255;
  long row = idx >> 8;
  const float* dr = dBC + row * 48;
  float s = bdt[c];
#pragma unroll
  for (int r = 0; r < DTRANK; ++r) s += dr[r] * Wdt[c * DTRANK + r];
  dlt[row * DI + c] = (s > 20.f) ? s : log1pf(__expf(s));
}

// ----------------------------- selective scan (wave32) ---------------------
__global__ __launch_bounds__(256) void scan_kernel(const float* __restrict__ dlt,
                                                   const float* __restrict__ xm,
                                                   const float* __restrict__ dBC,
                                                   const float* __restrict__ A_log,
                                                   const float* __restrict__ D_param,
                                                   float* __restrict__ y, int dir) {
  int tid  = threadIdx.x;
  int lane = tid & 31;
  int s    = lane & 15;
  int wg   = blockIdx.x * 8 + (tid >> 5);      // 0..511
  int cg   = wg * 2 + (lane >> 4);             // 0..1023
  int b = cg >> 8, d = cg & 255;
  float A  = -__expf(A_log[d * DSTATE + s]);
  float Dp = D_param[d];
  float h = 0.f;
  long base = (long)b * LSEQ;
  for (int t = 0; t < LSEQ; ++t) {
    int l = dir ? (LSEQ - 1 - t) : t;
    long row = base + l;
    if (t + 1 < LSEQ) {
      long nrow = base + (dir ? (LSEQ - 2 - t) : t + 1);
      __builtin_prefetch(&dlt[nrow * DI + d], 0, 1);
      __builtin_prefetch(&dBC[nrow * 48 + 8 + s], 0, 1);
    }
    float dv = dlt[row * DI + d];
    float xv = xm[row * DI + d];
    float Bv = dBC[row * 48 + 8 + s];
    float Cv = dBC[row * 48 + 24 + s];
    h = __expf(dv * A) * h + (dv * xv) * Bv;
    float c = h * Cv;
    c += __shfl_xor(c, 1, 16);
    c += __shfl_xor(c, 2, 16);
    c += __shfl_xor(c, 4, 16);
    c += __shfl_xor(c, 8, 16);
    if (s == 0) {
      float val = c + Dp * xv;
      if (dir) y[row * DI + d] += val;
      else     y[row * DI + d]  = val;
    }
  }
}

// ----------------------------- GEMM 3: out = (y*silu(z)) @ W_out^T ---------
__global__ __launch_bounds__(256) void gemm_out(const float* __restrict__ Y,
                                                const float* __restrict__ XZ,
                                                const unsigned short* __restrict__ Wb_,
                                                float* __restrict__ Out) {
  const __bf16* Wb = (const __bf16*)Wb_;
  __shared__ alignas(16) __bf16 As[64 * 32];
  __shared__ alignas(16) __bf16 Bs[128 * 32];
  int tid = threadIdx.x, wave = tid >> 5, lane = tid & 31;
  int mBlock = blockIdx.x;
  int mSub = (wave & 3) * 16;
  int nGrp = (wave >> 2) * 64;
  v8f acc[4] = {};
  for (int ks = 0; ks < 8; ++ks) {
    unsigned* Ap = (unsigned*)As;
    for (int i = tid; i < 64 * 16; i += 256) {
      int r = i >> 4, kk = (i & 15) * 2;
      int row = mBlock * 64 + r, col = ks * 32 + kk;
      const float2 zv = *(const float2*)&XZ[(long)row * (2 * DI) + DI + col];
      const float2 yv = *(const float2*)&Y[(long)row * DI + col];
      float g0 = yv.x * (zv.x / (1.f + __expf(-zv.x)));
      float g1 = yv.y * (zv.y / (1.f + __expf(-zv.y)));
      Ap[i] = pack_bf16(g0, g1);
    }
    async_copy_tile(Wb + ks * (128 * 32), Bs, 128 * 4, tid, 256);
    __syncthreads();
    v16bf a = load_a_frag(As, mSub, lane);
#pragma unroll
    for (int j = 0; j < 4; ++j) {
      v16bf b = load_b_frag(Bs, nGrp + j * 16, lane);
      acc[j] = __builtin_amdgcn_wmma_f32_16x16x32_bf16(
          false, a, false, b, (short)0, acc[j], false, false);
    }
    __syncthreads();
  }
#pragma unroll
  for (int j = 0; j < 4; ++j)
    store_c(Out, DM, mBlock * 64 + mSub, nGrp + j * 16, lane, acc[j]);
}

// ----------------------------- host orchestration --------------------------
extern "C" void kernel_launch(void* const* d_in, const int* in_sizes, int n_in,
                              void* d_out, int out_size, void* d_ws, size_t ws_size,
                              hipStream_t stream) {
  const float* x        = (const float*)d_in[0];
  const float* W_in     = (const float*)d_in[1];
  const float* conv1d_w = (const float*)d_in[2];
  const float* conv1d_b = (const float*)d_in[3];
  const float* W_xproj  = (const float*)d_in[4];
  const float* W_dt     = (const float*)d_in[5];
  const float* b_dt     = (const float*)d_in[6];
  const float* A_log    = (const float*)d_in[7];
  const float* D_param  = (const float*)d_in[8];
  const float* W_out    = (const float*)d_in[9];
  const float* conv1_w  = (const float*)d_in[10];
  const float* offset_w = (const float*)d_in[11];

  float* out    = (float*)d_out;                 // (B,L,128) = 1310720 floats
  float* offmap = out + (long)BATCH * LSEQ * DM; // (B,2,4,4) = 128 floats

  float* f   = (float*)d_ws;
  float* xo  = f;                                    // 16384
  float* xc1 = xo  + BATCH * 2 * DM * 16;            // 8192
  float* xr  = xc1 + BATCH * DM * 16;                // 1310720
  float* xz  = xr  + (long)BATCH * LSEQ * DM;        // 5242880
  float* xm  = xz  + (long)BATCH * LSEQ * 2 * DI;    // 2621440
  float* dBC = xm  + (long)BATCH * LSEQ * DI;        // 491520
  float* dlt = dBC + (long)BATCH * LSEQ * 48;        // 2621440
  float* yb  = dlt + (long)BATCH * LSEQ * DI;        // 2621440
  unsigned short* WinB  = (unsigned short*)(yb + (long)BATCH * LSEQ * DI);
  unsigned short* WxpB  = WinB + 2 * DI * DM;        // 65536
  unsigned short* WoutB = WxpB + 48 * DI;            // 12288

  convert_weights<<<(2 * DI * DM + 255) / 256, 256, 0, stream>>>(
      W_in, W_xproj, W_out, WinB, WxpB, WoutB);
  pool_kernel<<<(BATCH * DM * 16 + 255) / 256, 256, 0, stream>>>(x, xo);
  dwconv_kernel<<<(BATCH * DM * 16 + 255) / 256, 256, 0, stream>>>(xo, conv1_w, xc1);
  offconv_kernel<<<1, 128, 0, stream>>>(xc1, offset_w, offmap);
  warp_kernel<<<(int)(((long)BATCH * TT * 16 * DM + 255) / 256), 256, 0, stream>>>(
      x, offmap, xr);
  gemm_xin<<<dim3(BATCH * LSEQ / 64, (2 * DI) / 128), 256, 0, stream>>>(xr, WinB, xz);
  conv1d_kernel<<<(int)(((long)BATCH * LSEQ * DI + 255) / 256), 256, 0, stream>>>(
      xz, conv1d_w, conv1d_b, xm);
  gemm_xproj<<<BATCH * LSEQ / 64, 128, 0, stream>>>(xm, WxpB, dBC);
  dt_kernel<<<(int)(((long)BATCH * LSEQ * DI + 255) / 256), 256, 0, stream>>>(
      dBC, W_dt, b_dt, dlt);
  scan_kernel<<<64, 256, 0, stream>>>(dlt, xm, dBC, A_log, D_param, yb, 0);
  scan_kernel<<<64, 256, 0, stream>>>(dlt, xm, dBC, A_log, D_param, yb, 1);
  gemm_out<<<BATCH * LSEQ / 64, 256, 0, stream>>>(yb, xz, WoutB, out);

  (void)in_sizes; (void)n_in; (void)out_size; (void)ws_size;
}
